// MapAgent_33586644254746
// MI455X (gfx1250) — compile-verified, matrix operand order checked by reading, not run on hardware
//
#include <hip/hip_runtime.h>
#include <math.h>

// ---------------------------------------------------------------------------
// MapAgent forward on MI455X (gfx1250, wave32, WMMA)
//
// conv1/conv2/conv3/fc  -> implicit-GEMM with v_wmma_f32_16x16x32_f16
//                          (templated geometry; per-block LDS im2col offset
//                           table removes all div/mod from the hot loop;
//                           LDS tiles laid out so fragments load as b128)
// scan + big head GEMMs -> rank-1-column incremental update (done-masked)
// ---------------------------------------------------------------------------

typedef __attribute__((ext_vector_type(16))) _Float16 v16h;
typedef __attribute__((ext_vector_type(8)))  _Float16 v8h;
typedef __attribute__((ext_vector_type(8)))  float    v8f;

#define TB        1024      // T*B
#define MAP_DIM   517
#define GRID_HW   256       // 16*16
#define FLAT      132352    // 517*256
#define NACT      5

// ---------------------------------------------------------------------------
// Implicit-GEMM conv (VALID, NCHW / OIHW), fused bias + ReLU, geometry as
// template constants.  GEMM view: C[m,n] = sum_k im2col(in)[m,k] * w[n,k].
// Guarantees used (true for every instantiation below): M % 32 == 0,
// K % 32 == 0, Cout % 32 == 0  -> fully branchless, no bounds checks.
// Block: 128 threads = 4 waves; block tile 32x32 (2x2 waves of 16x16 WMMA).
// LDS: As[row][k], Bs[n][k] (operand-row-major, 16B-aligned 8-half groups so
// each lane's fragment = two ds_load_b128) + im2col offset table ofstab[K].
// ---------------------------------------------------------------------------
template <int CIN, int HIN, int WIN, int COUT,
          int KH, int KW, int STRIDE, int OH, int OW>
__global__ void __launch_bounds__(128)
conv_gemm_wmma(const float* __restrict__ in, const float* __restrict__ w,
               const float* __restrict__ bias, float* __restrict__ out)
{
    constexpr int OHW = OH * OW;
    constexpr int KHW = KH * KW;
    constexpr int K   = CIN * KHW;

    const int m0 = blockIdx.x * 32;
    const int n0 = blockIdx.y * 32;
    const int tid = threadIdx.x;

    __shared__ _Float16 As[32][40];   // [m-in-tile][k-in-tile], row stride 80B
    __shared__ _Float16 Bs[32][40];   // [n-in-tile][k-in-tile]
    __shared__ int ofstab[K];         // im2col column -> input element offset

    // build the k -> input-offset table once per block (decode leaves the loop)
    for (int k = tid; k < K; k += 128) {
        const int ci = k / KHW;
        const int rk = k - ci * KHW;
        const int ky = rk / KW;
        const int kx = rk - ky * KW;
        ofstab[k] = (ci * HIN + ky) * WIN + kx;
    }

    const int wid  = tid >> 5;            // wave 0..3
    const int lane = tid & 31;
    const int lr   = lane & 15;
    const int hi8  = (lane >> 4) << 3;    // 0 or 8: half-wave K-octet select
    const int warpM = (wid >> 1) << 4;    // 0 or 16
    const int warpN = (wid & 1) << 4;     // 0 or 16

    // staging coordinates: thread -> (row 0..31, 8 consecutive k)
    const int arow = tid >> 2;
    const int akg  = (tid & 3) << 3;

    // hoist im2col row decode (constant divisors, once per kernel)
    const int m   = m0 + arow;
    const int img = m / OHW;
    const int r2  = m - img * OHW;
    const int oy  = r2 / OW;
    const int ox  = r2 - oy * OW;
    const float* inrow = in + img * CIN * HIN * WIN
                            + (oy * STRIDE) * WIN + ox * STRIDE;
    const float* wrowp = w + (n0 + arow) * K;

    v8f acc = {};
    __syncthreads();                       // ofstab ready

    for (int k0 = 0; k0 < K; k0 += 32) {
        if (k0 + 32 < K) __builtin_prefetch(wrowp + k0 + 32, 0, 1);

        // ---- stage: gather 8 A elems + 8 B elems, cvt f32->f16, pack ----
        v8h av, bv;
#pragma unroll
        for (int j = 0; j < 8; ++j) {
            av[j] = (_Float16)inrow[ofstab[k0 + akg + j]];
            bv[j] = (_Float16)wrowp[k0 + akg + j];
        }
        *(v8h*)&As[arow][akg] = av;   // one ds_store_b128
        *(v8h*)&Bs[arow][akg] = bv;   // one ds_store_b128
        __syncthreads();

        // ---- fragments: two contiguous b128 LDS loads per operand ----
        const v8h a_lo = *(const v8h*)&As[warpM + lr][hi8];
        const v8h a_hi = *(const v8h*)&As[warpM + lr][16 + hi8];
        const v8h b_lo = *(const v8h*)&Bs[warpN + lr][hi8];
        const v8h b_hi = *(const v8h*)&Bs[warpN + lr][16 + hi8];
        const v16h va = __builtin_shufflevector(a_lo, a_hi,
                            0,1,2,3,4,5,6,7,8,9,10,11,12,13,14,15);
        const v16h vb = __builtin_shufflevector(b_lo, b_hi,
                            0,1,2,3,4,5,6,7,8,9,10,11,12,13,14,15);
        acc = __builtin_amdgcn_wmma_f32_16x16x32_f16(
                  false, va, false, vb, (short)0, acc, false, false);
        __syncthreads();
    }

    // ---- epilogue: C layout — VGPR j: lanes 0-15 -> M=j, lanes 16-31 -> M=j+8
    const int n  = n0 + warpN + lr;
    const float bn = bias[n];
#pragma unroll
    for (int j = 0; j < 8; ++j) {
        const int mm  = m0 + warpM + hi8 + j;
        float v = acc[j] + bn;
        v = v > 0.f ? v : 0.f;                    // ReLU
        const int im2 = mm / OHW;                 // constant divisor
        const int rr  = mm - im2 * OHW;
        const int yy  = rr / OW;                  // constant divisor
        const int xx  = rr - yy * OW;
        out[((im2 * COUT + n) * OH + yy) * OW + xx] = v;
    }
}

// ---------------------------------------------------------------------------
// h = concat(feat(1024x512), onehot(last_action, 5))   -> (1024, 517)
// ---------------------------------------------------------------------------
__global__ void build_h_kernel(const float* __restrict__ feat,
                               const int* __restrict__ last_action,
                               float* __restrict__ h)
{
    int idx = blockIdx.x * blockDim.x + threadIdx.x;
    if (idx >= TB * MAP_DIM) return;
    const int i = idx / MAP_DIM;
    const int j = idx - i * MAP_DIM;
    float v;
    if (j < 512) v = feat[i * 512 + j];
    else         v = (last_action[i] == (j - 512)) ? 1.f : 0.f;
    h[idx] = v;
}

// ---------------------------------------------------------------------------
// Sequential scan with incremental head update. One block per env (b).
// state lives directly in d_out's final_state region.
//   y_t = (1-d)*y_{t-1} + Wcol(p).h_t - (1-d)*Wcol(p).state_col(p)
// 256 threads = 128 (head,neuron) pairs x 2 K-partitions.
// ---------------------------------------------------------------------------
__global__ void __launch_bounds__(256)
scan_kernel(const float* __restrict__ h, const float* __restrict__ done,
            const int* __restrict__ position, const float* __restrict__ state0,
            const float* __restrict__ wpol, const float* __restrict__ wval,
            float* __restrict__ ypolAll, float* __restrict__ yvalAll,
            float* __restrict__ stateOut)
{
    const int b   = blockIdx.x;
    const int tid = threadIdx.x;
    float* st = stateOut + b * FLAT;

    // init state from state0
    for (int k = tid; k < FLAT; k += 256) st[k] = state0[b * FLAT + k];

    __shared__ float hcol[MAP_DIM];
    __shared__ float scol[MAP_DIM];
    __shared__ float ybuf[128];
    __shared__ float red[256];

    const int nh   = tid & 127;      // (head, neuron)
    const int head = nh >> 6;
    const int nn   = nh & 63;
    const int q    = tid >> 7;       // K partition 0/1
    const float* wrow = (head ? wval : wpol) + nn * FLAT;
    __syncthreads();

    // y0 = wrow . state0[b]  (state0 may be nonzero in general)
    {
        float acc = 0.f;
        const int kb = q * (FLAT / 2), ke = kb + (FLAT / 2);
        const float* sb = state0 + b * FLAT;
        for (int k = kb; k < ke; ++k) acc += wrow[k] * sb[k];
        red[tid] = acc;
        __syncthreads();
        if (q == 0) ybuf[nh] = red[tid] + red[tid + 128];
        __syncthreads();
    }

    for (int t = 0; t < 32; ++t) {
        const int i = t * 32 + b;
        const float d = done[i];
        const float scale = 1.f - d;
        const float* hr = h + i * MAP_DIM;
        const int r = position[i * 2 + 0];
        const int c = position[i * 2 + 1];
        const int p = r * 16 + c;

        for (int m = tid; m < MAP_DIM; m += 256) {
            hcol[m] = hr[m];
            scol[m] = st[m * GRID_HW + p];
        }
        __syncthreads();

        // two dots per (head,neuron): old column & new column
        float accO = 0.f, accN = 0.f;
        const int mb = q ? 259 : 0;
        const int me = q ? MAP_DIM : 259;
        for (int m = mb; m < me; ++m) {
            const float wv = wrow[m * GRID_HW + p];
            accO += wv * scol[m];
            accN += wv * hcol[m];
        }
        red[tid] = accN - scale * accO;
        __syncthreads();
        if (q == 0) {
            const float ynew = scale * ybuf[nh] + red[tid] + red[tid + 128];
            ybuf[nh] = ynew;
            (head ? yvalAll : ypolAll)[i * 64 + nn] = ynew;
        }
        __syncthreads();

        // state <- (1-d)*state  (skip when d==0: the common case), then col p = h
        if (d != 0.f) {
            for (int k = tid; k < FLAT; k += 256) st[k] *= scale;
            __syncthreads();
        }
        for (int m = tid; m < MAP_DIM; m += 256) st[m * GRID_HW + p] = hcol[m];
        __syncthreads();
    }
}

// ---------------------------------------------------------------------------
// Final heads: logits = tanh(ypol+b1) @ pol_w2^T + b2 ; v likewise.
// ---------------------------------------------------------------------------
__global__ void heads_kernel(const float* __restrict__ ypolAll,
                             const float* __restrict__ yvalAll,
                             const float* __restrict__ pol_b1,
                             const float* __restrict__ pol_w2,
                             const float* __restrict__ pol_b2,
                             const float* __restrict__ val_b1,
                             const float* __restrict__ val_w2,
                             const float* __restrict__ val_b2,
                             float* __restrict__ logits,
                             float* __restrict__ vout)
{
    const int i = blockIdx.x * blockDim.x + threadIdx.x;
    if (i >= TB) return;
    float lg[NACT];
#pragma unroll
    for (int a = 0; a < NACT; ++a) lg[a] = pol_b2[a];
    float vv = val_b2[0];
    for (int n = 0; n < 64; ++n) {
        const float tp = tanhf(ypolAll[i * 64 + n] + pol_b1[n]);
        const float tv = tanhf(yvalAll[i * 64 + n] + val_b1[n]);
#pragma unroll
        for (int a = 0; a < NACT; ++a) lg[a] += tp * pol_w2[a * 64 + n];
        vv += tv * val_w2[n];
    }
#pragma unroll
    for (int a = 0; a < NACT; ++a) logits[i * NACT + a] = lg[a];
    vout[i] = vv;
}

// ---------------------------------------------------------------------------
extern "C" void kernel_launch(void* const* d_in, const int* in_sizes, int n_in,
                              void* d_out, int out_size, void* d_ws, size_t ws_size,
                              hipStream_t stream)
{
    const float* image       = (const float*)d_in[0];
    const int*   last_action = (const int*)  d_in[1];
    const int*   position    = (const int*)  d_in[2];
    const float* done        = (const float*)d_in[3];
    const float* state0      = (const float*)d_in[4];
    const float* c1_w = (const float*)d_in[5];
    const float* c1_b = (const float*)d_in[6];
    const float* c2_w = (const float*)d_in[7];
    const float* c2_b = (const float*)d_in[8];
    const float* c3_w = (const float*)d_in[9];
    const float* c3_b = (const float*)d_in[10];
    const float* fc_w = (const float*)d_in[11];
    const float* fc_b = (const float*)d_in[12];
    const float* pol_w1 = (const float*)d_in[13];
    const float* pol_b1 = (const float*)d_in[14];
    const float* pol_w2 = (const float*)d_in[15];
    const float* pol_b2 = (const float*)d_in[16];
    const float* val_w1 = (const float*)d_in[17];
    const float* val_b1 = (const float*)d_in[18];
    const float* val_w2 = (const float*)d_in[19];
    const float* val_b2 = (const float*)d_in[20];

    float* ws = (float*)d_ws;
    float* x1      = ws;                        // 1024*32*15*15
    float* x2      = x1 + 1024 * 32 * 15 * 15;  // 1024*64*6*6
    float* x3      = x2 + 1024 * 64 * 6 * 6;    // 1024*64*4*4 (= fc input 1024x1024)
    float* feat    = x3 + 1024 * 1024;          // 1024*512
    float* h       = feat + 1024 * 512;         // 1024*517
    float* ypolAll = h + TB * MAP_DIM;          // 1024*64
    float* yvalAll = ypolAll + TB * 64;         // 1024*64

    float* logits    = (float*)d_out;           // 1024*5
    float* vout      = logits + TB * NACT;      // 1024
    float* final_st  = vout + TB;               // 32*517*256 (also scan's state)

    // conv1: (1024,3,64,64) -> (1024,32,15,15)   M=230400 K=192
    conv_gemm_wmma<3, 64, 64, 32, 8, 8, 4, 15, 15>
        <<<dim3(230400 / 32, 1), 128, 0, stream>>>(image, c1_w, c1_b, x1);
    // conv2: -> (1024,64,6,6)                    M=36864  K=512
    conv_gemm_wmma<32, 15, 15, 64, 4, 4, 2, 6, 6>
        <<<dim3(36864 / 32, 2), 128, 0, stream>>>(x1, c2_w, c2_b, x2);
    // conv3: -> (1024,64,4,4)                    M=16384  K=576
    conv_gemm_wmma<64, 6, 6, 64, 3, 3, 1, 4, 4>
        <<<dim3(16384 / 32, 2), 128, 0, stream>>>(x2, c3_w, c3_b, x3);
    // fc as 1x1 conv: (1024,1024,1,1) -> (1024,512,1,1)   M=1024 K=1024
    conv_gemm_wmma<1024, 1, 1, 512, 1, 1, 1, 1, 1>
        <<<dim3(1024 / 32, 512 / 32), 128, 0, stream>>>(x3, fc_w, fc_b, feat);

    build_h_kernel<<<(TB * MAP_DIM + 255) / 256, 256, 0, stream>>>(feat, last_action, h);

    scan_kernel<<<32, 256, 0, stream>>>(h, done, position, state0,
                                        pol_w1, val_w1, ypolAll, yvalAll, final_st);

    heads_kernel<<<(TB + 255) / 256, 256, 0, stream>>>(
        ypolAll, yvalAll, pol_b1, pol_w2, pol_b2, val_b1, val_w2, val_b2,
        logits, vout);
}
